// ModernRelationBlock_14216341749898
// MI455X (gfx1250) — matrix operand order repaired and checked by reading, not compiled, not run on hardware
//
#include <hip/hip_runtime.h>
#include <hip/hip_fp16.h>

typedef __attribute__((ext_vector_type(16))) _Float16 v16h;
typedef __attribute__((ext_vector_type(8)))  float    v8f;
typedef _Float16 half_t;

#define NN 50000
#define EE 600000
#define HH 128
#define EDD 64

// ---------------------------------------------------------------- helpers

__device__ __forceinline__ float gelu_exact(float x) {
    return 0.5f * x * (1.0f + erff(x * 0.70710678118654752f));
}

__device__ __forceinline__ v8f wmma32(v16h a, v16h b, v8f c) {
    // D = A(16x32 f16) * B(32x16 f16) + C(16x16 f32)
    return __builtin_amdgcn_wmma_f32_16x16x32_f16(
        /*neg_a=*/false, a, /*neg_b=*/false, b,
        /*c_mod=*/(short)0, c, /*reuse_a=*/false, /*reuse_b=*/false);
}

// A fragment: 16x32 f16 tile, row-major source with leading dim ldk (halves).
// lane<16: M=lane, K = k0+[0..7], k0+16+[0..7]; lane>=16: +8 on K groups.
__device__ __forceinline__ v16h load_A_frag(const half_t* base, int ldk, int k0, int lane) {
    int m    = lane & 15;
    int koff = (lane >> 4) << 3;
    const half_t* p = base + m * ldk + k0 + koff;
    union { uint4 u[2]; v16h h; } cvt;
    cvt.u[0] = *(const uint4*)(p);
    cvt.u[1] = *(const uint4*)(p + 16);
    return cvt.h;
}

// B fragment from pre-packed weights: fragment (kt,nt) is 32 lanes x 16 halves contiguous.
__device__ __forceinline__ v16h load_B_frag(const half_t* packed, int NT, int kt, int nt, int lane) {
    const half_t* p = packed + ((size_t)((kt * NT + nt) * 32 + lane)) * 16;
    union { uint4 u[2]; v16h h; } cvt;
    cvt.u[0] = *(const uint4*)(p);
    cvt.u[1] = *(const uint4*)(p + 8);
    return cvt.h;
}

// ---------------------------------------------------------------- utility kernels

__global__ void k_zero(float* __restrict__ agg, float* __restrict__ deg) {
    int stride = gridDim.x * blockDim.x;
    for (int i = blockIdx.x * blockDim.x + threadIdx.x; i < NN * HH; i += stride) agg[i] = 0.0f;
    for (int j = blockIdx.x * blockDim.x + threadIdx.x; j < NN; j += stride) deg[j] = 0.0f;
}

// Pack row-major [K x Ncols] f32 weight into WMMA B-fragment f16 layout.
// out[((kt*NT+nt)*32 + lane)*16 + j] = W[(kt*32 + (j>>3)*16 + (lane>>4)*8 + (j&7))*Ncols + nt*16 + (lane&15)]
__global__ void k_pack_b(const float* __restrict__ W, half_t* __restrict__ out, int K, int Ncols) {
    int NT  = Ncols >> 4;
    int tot = K * Ncols;
    for (int o = blockIdx.x * blockDim.x + threadIdx.x; o < tot; o += gridDim.x * blockDim.x) {
        int j    = o & 15;
        int lane = (o >> 4) & 31;
        int tile = o >> 9;
        int nt   = tile % NT;
        int kt   = tile / NT;
        int krow = (kt << 5) + ((j >> 3) << 4) + ((lane >> 4) << 3) + (j & 7);
        int ncol = (nt << 4) + (lane & 15);
        out[o] = (half_t)W[(size_t)krow * Ncols + ncol];
    }
}

// LayerNorm over H=128, one row per block (128 threads), f16 output.
__global__ __launch_bounds__(128)
void k_layernorm_f16(const float* __restrict__ x, const float* __restrict__ g,
                     const float* __restrict__ b, half_t* __restrict__ out) {
    __shared__ float red[128];
    const int row = blockIdx.x;
    const int t   = threadIdx.x;
    float v = x[(size_t)row * HH + t];
    red[t] = v; __syncthreads();
    for (int s = 64; s > 0; s >>= 1) { if (t < s) red[t] += red[t + s]; __syncthreads(); }
    float mu = red[0] * (1.0f / HH);
    __syncthreads();
    float d = v - mu;
    red[t] = d * d; __syncthreads();
    for (int s = 64; s > 0; s >>= 1) { if (t < s) red[t] += red[t + s]; __syncthreads(); }
    float rs = rsqrtf(red[0] * (1.0f / HH) + 1e-5f);
    out[(size_t)row * HH + t] = (half_t)(d * rs * g[t] + b[t]);
}

// ---------------------------------------------------------------- edge kernel

__global__ __launch_bounds__(256)
void k_edge(const half_t* __restrict__ h16,
            const int* __restrict__ esrc, const int* __restrict__ edst,
            const float* __restrict__ eemb,
            const half_t* __restrict__ Wm1p, const float* __restrict__ bm1,
            const half_t* __restrict__ Wm2p, const float* __restrict__ bm2,
            const half_t* __restrict__ Wg1p, const float* __restrict__ bg1,
            const float* __restrict__ Wg2,  const float* __restrict__ bg2,
            float* __restrict__ agg, float* __restrict__ deg,
            float* __restrict__ edge_repr) {
    __shared__ alignas(16) half_t Ain[16 * 320];   // [edge][ h_dst(128) | h_src(128) | e(64) ]
    __shared__ alignas(16) half_t msgh[16 * 128];  // gelu(msg hidden), f16
    __shared__ float  gh[16 * 128];                // gelu(gate hidden), f32
    __shared__ float  gate_s[16];
    __shared__ int    dst_s[16];
    __shared__ int    src_s[16];

    const int t    = threadIdx.x;
    const int lane = t & 31;
    const int w    = t >> 5;          // wave id 0..7 -> output column tile
    const int e0   = blockIdx.x * 16;

    if (t < 16) { dst_s[t] = edst[e0 + t]; src_s[t] = esrc[e0 + t]; }
    __syncthreads();

    // Stage gate input [16 x 320]; msg input is its column slice [:,128:320].
    for (int c4 = t; c4 < 16 * 80; c4 += 256) {
        int i = c4 / 80;
        int c = (c4 % 80) * 4;
        half_t v0, v1, v2, v3;
        if (c < 128) {
            const half_t* p = h16 + (size_t)dst_s[i] * HH + c;
            v0 = p[0]; v1 = p[1]; v2 = p[2]; v3 = p[3];
        } else if (c < 256) {
            const half_t* p = h16 + (size_t)src_s[i] * HH + (c - 128);
            v0 = p[0]; v1 = p[1]; v2 = p[2]; v3 = p[3];
        } else {
            const float* p = eemb + (size_t)(e0 + i) * EDD + (c - 256);
            v0 = (half_t)p[0]; v1 = (half_t)p[1]; v2 = (half_t)p[2]; v3 = (half_t)p[3];
        }
        half_t* q = Ain + i * 320 + c;
        q[0] = v0; q[1] = v1; q[2] = v2; q[3] = v3;
    }
    __syncthreads();

    const int n0    = w << 4;
    const int nn    = n0 + (lane & 15);
    const int mbase = (lane >> 4) << 3;

    // gate hidden: [16x320] @ W_g1[320x128], K = 10 x 32
    v8f cg = {0.f, 0.f, 0.f, 0.f, 0.f, 0.f, 0.f, 0.f};
#pragma unroll
    for (int kt = 0; kt < 10; ++kt)
        cg = wmma32(load_A_frag(Ain, 320, kt * 32, lane),
                    load_B_frag(Wg1p, 8, kt, w, lane), cg);

    // msg hidden: [16x192] @ W_m1[192x128], K = 6 x 32
    v8f cm = {0.f, 0.f, 0.f, 0.f, 0.f, 0.f, 0.f, 0.f};
#pragma unroll
    for (int kt = 0; kt < 6; ++kt)
        cm = wmma32(load_A_frag(Ain + 128, 320, kt * 32, lane),
                    load_B_frag(Wm1p, 8, kt, w, lane), cm);

    float bgv = bg1[nn], bmv = bm1[nn];
#pragma unroll
    for (int r = 0; r < 8; ++r) {
        int m = mbase + r;                 // C/D layout: reg r, lanes>=16 -> M += 8
        gh[m * 128 + nn]   = gelu_exact(cg[r] + bgv);
        msgh[m * 128 + nn] = (half_t)gelu_exact(cm[r] + bmv);
    }
    __syncthreads();

    // gate = sigmoid(gh @ W_g2 + b_g2)  (H->1: VALU dot, not worth a matrix op)
    if (t < 16) {
        float acc = bg2[0];
        for (int k = 0; k < 128; ++k) acc += gh[t * 128 + k] * Wg2[k];
        gate_s[t] = 1.0f / (1.0f + __expf(-acc));
    }
    __syncthreads();

    // msg = msgh @ W_m2[128x128], K = 4 x 32
    v8f c2 = {0.f, 0.f, 0.f, 0.f, 0.f, 0.f, 0.f, 0.f};
#pragma unroll
    for (int kt = 0; kt < 4; ++kt)
        c2 = wmma32(load_A_frag(msgh, 128, kt * 32, lane),
                    load_B_frag(Wm2p, 8, kt, w, lane), c2);

    float bb = bm2[nn];
#pragma unroll
    for (int r = 0; r < 8; ++r) {
        int m = mbase + r;
        float val = (c2[r] + bb) * gate_s[m];
        edge_repr[(size_t)(e0 + m) * HH + nn] = val;
        atomicAdd(&agg[(size_t)dst_s[m] * HH + nn], val);
    }
    if (t < 16) atomicAdd(&deg[dst_s[t]], 1.0f);
}

// ---------------------------------------------------------------- node update

__global__ __launch_bounds__(256)
void k_node(const float* __restrict__ x,
            const half_t* __restrict__ h16,
            const float* __restrict__ agg, const float* __restrict__ deg,
            const half_t* __restrict__ Wselfp, const float* __restrict__ bself,
            const half_t* __restrict__ Waggp,  const float* __restrict__ bagg,
            float* __restrict__ out) {
    __shared__ alignas(16) half_t aggn[16 * 128];
    const int t = threadIdx.x, lane = t & 31, w = t >> 5;
    const int r0 = blockIdx.x * 16;

    for (int idx = t; idx < 16 * 128; idx += 256) {
        int i = idx >> 7, n = idx & 127;
        float d = deg[r0 + i];
        aggn[idx] = (half_t)(agg[(size_t)(r0 + i) * HH + n] / fmaxf(d, 1.0f));
    }
    __syncthreads();

    v8f c = {0.f, 0.f, 0.f, 0.f, 0.f, 0.f, 0.f, 0.f};
    const half_t* Abase = h16 + (size_t)r0 * HH;
#pragma unroll
    for (int kt = 0; kt < 4; ++kt)
        c = wmma32(load_A_frag(Abase, 128, kt * 32, lane),
                   load_B_frag(Wselfp, 8, kt, w, lane), c);
#pragma unroll
    for (int kt = 0; kt < 4; ++kt)
        c = wmma32(load_A_frag(aggn, 128, kt * 32, lane),
                   load_B_frag(Waggp, 8, kt, w, lane), c);

    const int nn = (w << 4) + (lane & 15);
    const int mbase = (lane >> 4) << 3;
    float bsum = bself[nn] + bagg[nn];
#pragma unroll
    for (int r = 0; r < 8; ++r) {
        int row = r0 + mbase + r;
        out[(size_t)row * HH + nn] = x[(size_t)row * HH + nn] + c[r] + bsum;
    }
}

// ---------------------------------------------------------------- FFN

__global__ __launch_bounds__(256)
void k_ffn(const half_t* __restrict__ ffn16,
           const half_t* __restrict__ Wf1p, const float* __restrict__ bf1,
           const half_t* __restrict__ Wf2p, const float* __restrict__ bf2,
           float* __restrict__ out) {
    __shared__ alignas(16) half_t hid[16 * 256];
    const int t = threadIdx.x, lane = t & 31, w = t >> 5;
    const int r0 = blockIdx.x * 16;
    const half_t* Abase = ffn16 + (size_t)r0 * HH;
    const int mbase = (lane >> 4) << 3;

    // up: [16x128] @ W_f1[128x256]; 8 waves cover 16 N-tiles in two passes
#pragma unroll
    for (int pass = 0; pass < 2; ++pass) {
        int nt = w + pass * 8;
        v8f c = {0.f, 0.f, 0.f, 0.f, 0.f, 0.f, 0.f, 0.f};
#pragma unroll
        for (int kt = 0; kt < 4; ++kt)
            c = wmma32(load_A_frag(Abase, 128, kt * 32, lane),
                       load_B_frag(Wf1p, 16, kt, nt, lane), c);
        int nn = (nt << 4) + (lane & 15);
        float bv = bf1[nn];
#pragma unroll
        for (int r = 0; r < 8; ++r)
            hid[(mbase + r) * 256 + nn] = (half_t)gelu_exact(c[r] + bv);
    }
    __syncthreads();

    // down: [16x256] @ W_f2[256x128], K = 8 x 32
    v8f c = {0.f, 0.f, 0.f, 0.f, 0.f, 0.f, 0.f, 0.f};
#pragma unroll
    for (int kt = 0; kt < 8; ++kt)
        c = wmma32(load_A_frag(hid, 256, kt * 32, lane),
                   load_B_frag(Wf2p, 8, kt, w, lane), c);

    const int nn = (w << 4) + (lane & 15);
    float bv = bf2[nn];
#pragma unroll
    for (int r = 0; r < 8; ++r) {
        int row = r0 + mbase + r;
        out[(size_t)row * HH + nn] += c[r] + bv;  // residual
    }
}

// ---------------------------------------------------------------- launch

extern "C" void kernel_launch(void* const* d_in, const int* in_sizes, int n_in,
                              void* d_out, int out_size, void* d_ws, size_t ws_size,
                              hipStream_t stream) {
    (void)in_sizes; (void)n_in; (void)out_size; (void)ws_size;

    const float* x      = (const float*)d_in[0];
    const int*   esrc   = (const int*)  d_in[1];
    const int*   edst   = (const int*)  d_in[2];
    const float* eemb   = (const float*)d_in[3];
    const float* ln1_g  = (const float*)d_in[4];
    const float* ln1_b  = (const float*)d_in[5];
    const float* W_self = (const float*)d_in[6];
    const float* b_self = (const float*)d_in[7];
    const float* W_m1   = (const float*)d_in[8];
    const float* b_m1   = (const float*)d_in[9];
    const float* W_m2   = (const float*)d_in[10];
    const float* b_m2   = (const float*)d_in[11];
    const float* W_g1   = (const float*)d_in[12];
    const float* b_g1   = (const float*)d_in[13];
    const float* W_g2   = (const float*)d_in[14];
    const float* b_g2   = (const float*)d_in[15];
    const float* W_agg  = (const float*)d_in[16];
    const float* b_agg  = (const float*)d_in[17];
    const float* ln2_g  = (const float*)d_in[18];
    const float* ln2_b  = (const float*)d_in[19];
    const float* W_f1   = (const float*)d_in[20];
    const float* b_f1   = (const float*)d_in[21];
    const float* W_f2   = (const float*)d_in[22];
    const float* b_f2   = (const float*)d_in[23];

    float* out       = (float*)d_out;                 // [N,H]
    float* edge_repr = out + (size_t)NN * HH;         // [E,H]

    // workspace carve-out (256B aligned slices)
    char*  ws  = (char*)d_ws;
    size_t off = 0;
    auto carve = [&](size_t bytes) -> void* {
        void* p = ws + off;
        off += (bytes + 255) & ~(size_t)255;
        return p;
    };
    half_t* h16    = (half_t*)carve((size_t)NN * HH * 2);
    half_t* ffn16  = (half_t*)carve((size_t)NN * HH * 2);
    float*  agg    = (float*) carve((size_t)NN * HH * 4);
    float*  deg    = (float*) carve((size_t)NN * 4);
    half_t* Wm1p   = (half_t*)carve((size_t)192 * 128 * 2);
    half_t* Wm2p   = (half_t*)carve((size_t)128 * 128 * 2);
    half_t* Wg1p   = (half_t*)carve((size_t)320 * 128 * 2);
    half_t* Wselfp = (half_t*)carve((size_t)128 * 128 * 2);
    half_t* Waggp  = (half_t*)carve((size_t)128 * 128 * 2);
    half_t* Wf1p   = (half_t*)carve((size_t)128 * 256 * 2);
    half_t* Wf2p   = (half_t*)carve((size_t)256 * 128 * 2);

    // 0) zero scatter accumulators
    k_zero<<<512, 256, 0, stream>>>(agg, deg);

    // 1) pack weights into WMMA B-fragment layout (f16)
    k_pack_b<<<96,  256, 0, stream>>>(W_m1,   Wm1p,   192, 128);
    k_pack_b<<<64,  256, 0, stream>>>(W_m2,   Wm2p,   128, 128);
    k_pack_b<<<160, 256, 0, stream>>>(W_g1,   Wg1p,   320, 128);
    k_pack_b<<<64,  256, 0, stream>>>(W_self, Wselfp, 128, 128);
    k_pack_b<<<64,  256, 0, stream>>>(W_agg,  Waggp,  128, 128);
    k_pack_b<<<128, 256, 0, stream>>>(W_f1,   Wf1p,   128, 256);
    k_pack_b<<<128, 256, 0, stream>>>(W_f2,   Wf2p,   256, 128);

    // 2) LN1 -> h16
    k_layernorm_f16<<<NN, 128, 0, stream>>>(x, ln1_g, ln1_b, h16);

    // 3) per-edge MLPs + gated scatter (the 98-GFLOP bulk, WMMA)
    k_edge<<<EE / 16, 256, 0, stream>>>(h16, esrc, edst, eemb,
                                        Wm1p, b_m1, Wm2p, b_m2,
                                        Wg1p, b_g1, W_g2, b_g2,
                                        agg, deg, edge_repr);

    // 4) node update: out = x + h@W_self + (agg/deg)@W_agg
    k_node<<<NN / 16, 256, 0, stream>>>(x, h16, agg, deg,
                                        Wselfp, b_self, Waggp, b_agg, out);

    // 5) LN2 -> ffn16, 6) FFN with residual into out
    k_layernorm_f16<<<NN, 128, 0, stream>>>(out, ln2_g, ln2_b, ffn16);
    k_ffn<<<NN / 16, 256, 0, stream>>>(ffn16, Wf1p, b_f1, Wf2p, b_f2, out);
}
